// GNNDecoder_68143951118636
// MI455X (gfx1250) — matrix operand order repaired and checked by reading, not compiled, not run on hardware
//
#include <hip/hip_runtime.h>
#include <hip/hip_bf16.h>
#include <stdint.h>

// GNN decoder for MI455X (gfx1250).
//  - GEMMs: bf16 WMMA path (v_wmma_f32_16x16x32_bf16, f32 accumulate)
//  - Edge aggregation: gather via fixed-capacity adjacency (in-degree <= 4 on
//    this grid graph) fused with self-loop/bias + LayerNorm + ReLU, no atomics.

namespace {

constexpr int kB    = 32;
constexpr int kN    = 4096;
constexpr int kH    = 256;
constexpr int kOut  = 256;
constexpr int kL    = 4;
constexpr int kTotal = kB * kN;        // 131072 nodes
constexpr int kCap  = 6;               // adjacency slots per node (grid: <=4)
constexpr float kEps = 1e-5f;

typedef __bf16 v16bf __attribute__((ext_vector_type(16)));
typedef float  v8f   __attribute__((ext_vector_type(8)));

union FragAB { uint4 u[2]; v16bf v; };

__device__ __forceinline__ unsigned short f2bf(float x) {
  unsigned int u = __float_as_uint(x);
  u += 0x7fffu + ((u >> 16) & 1u);       // round-to-nearest-even
  return (unsigned short)(u >> 16);
}
__device__ __forceinline__ float bf2f(unsigned short h) {
  return __uint_as_float(((unsigned int)h) << 16);
}

// ---------------- initial projection: h = x * proj_w + proj_b (bf16) -------
__global__ void k_proj(const float* __restrict__ x, const float* __restrict__ pw,
                       const float* __restrict__ pb, unsigned short* __restrict__ hb) {
  int idx = blockIdx.x * blockDim.x + threadIdx.x;   // over kTotal*kH/4
  int node = idx >> 6;
  int f4 = (idx & 63) << 2;
  float xv = x[node];
  float4 w = *(const float4*)(pw + f4);
  float4 b = *(const float4*)(pb + f4);
  ushort4 o;
  o.x = f2bf(fmaf(xv, w.x, b.x));
  o.y = f2bf(fmaf(xv, w.y, b.y));
  o.z = f2bf(fmaf(xv, w.z, b.z));
  o.w = f2bf(fmaf(xv, w.w, b.w));
  ((ushort4*)hb)[idx] = o;
}

// ---------------- degree / normalization (per-graph, shared over batch) ----
__global__ void k_deg_init(float* __restrict__ deg, int* __restrict__ cnt) {
  int n = blockIdx.x * blockDim.x + threadIdx.x;
  deg[n] = 1.0f;   // self loop
  cnt[n] = 0;
}
__global__ void k_deg_acc(const int* __restrict__ dst, int E0, float* __restrict__ deg) {
  int e = blockIdx.x * blockDim.x + threadIdx.x;
  if (e < E0) atomicAdd(&deg[dst[e]], 1.0f);
}
__global__ void k_deg_fin(const float* __restrict__ deg, float* __restrict__ dinv,
                          float* __restrict__ selfn) {
  int n = blockIdx.x * blockDim.x + threadIdx.x;
  float d = deg[n];
  dinv[n]  = rsqrtf(d);
  selfn[n] = 1.0f / d;
}
// fixed-capacity in-adjacency: adj[n*kCap+j] = src, adjw = dinv[s]*dinv[d]
__global__ void k_adj_build(const int* __restrict__ src, const int* __restrict__ dst,
                            const float* __restrict__ dinv, int E0,
                            int* __restrict__ cnt, int* __restrict__ adj,
                            float* __restrict__ adjw) {
  int e = blockIdx.x * blockDim.x + threadIdx.x;
  if (e >= E0) return;
  int s = src[e], d = dst[e];
  int slot = atomicAdd(&cnt[d], 1);
  if (slot < kCap) {
    adj[d * kCap + slot]  = s;
    adjw[d * kCap + slot] = dinv[s] * dinv[d];
  }
}

// ---------------- pack conv_w[l] into WMMA B-fragment bf16 layout ----------
// pw layout: [nt(16)][kb(8)][lane(32)][16 halves], lane = N-col | K-half group
__global__ void k_pack_w(const float* __restrict__ W, unsigned short* __restrict__ pw) {
  int idx = blockIdx.x * blockDim.x + threadIdx.x;   // 65536
  int j    = idx & 15;
  int lane = (idx >> 4) & 31;
  int kb   = (idx >> 9) & 7;
  int nt   = idx >> 12;
  int n = (nt << 4) + (lane & 15);
  int k = (kb << 5) + ((j >> 3) << 4) + ((lane >> 4) << 3) + (j & 7);  // lanes 0-15: K 0-7,16-23; 16-31: K 8-15,24-31
  pw[idx] = f2bf(W[k * kH + n]);
}

// ---------------- WMMA GEMM: t[131072,256] = hb @ Wl -----------------------
// One wave computes a 16x64 output strip (4 N-tiles reuse one A fragment).
__global__ void k_gemm(const unsigned short* __restrict__ hb,
                       const unsigned short* __restrict__ pw,
                       float* __restrict__ t) {
  int gt   = blockIdx.x * blockDim.x + threadIdx.x;
  int lane = gt & 31;
  int wave = gt >> 5;
  int mt = wave >> 2;            // M tile (16 rows)
  int ng = wave & 3;             // which group of 4 N-tiles
  int row = (mt << 4) + (lane & 15);
  int sh  = (lane >> 4) << 3;    // A layout K-offset: 0 (lanes 0-15) or 8

  const uint4* arow  = (const uint4*)(hb + (size_t)row * kH);  // uint4 = 8 halves
  const uint4* bbase = (const uint4*)pw;

  v8f zero = {0.f,0.f,0.f,0.f,0.f,0.f,0.f,0.f};
  v8f c[4];
#pragma unroll
  for (int i = 0; i < 4; ++i) c[i] = zero;

  for (int kb = 0; kb < 8; ++kb) {                    // K = 256 in steps of 32
    FragAB a;
    a.u[0] = arow[kb * 4 + (sh >> 3)];                // K = kb*32 + sh .. +7
    a.u[1] = arow[kb * 4 + 2 + (sh >> 3)];            // K = kb*32 + 16 + sh .. +7
#pragma unroll
    for (int i = 0; i < 4; ++i) {
      int nt = (ng << 2) + i;
      const uint4* bp = bbase + ((((nt * 8 + kb) << 5) + lane) << 1);
      FragAB b;
      b.u[0] = bp[0];
      b.u[1] = bp[1];
      c[i] = __builtin_amdgcn_wmma_f32_16x16x32_bf16(
          false, a.v, false, b.v, (short)0, c[i], false, false);
    }
  }

  int mrow = (mt << 4) + ((lane >> 4) << 3);
  int col  = lane & 15;
#pragma unroll
  for (int i = 0; i < 4; ++i) {
    int ncol = ((((ng << 2) + i)) << 4) + col;
#pragma unroll
    for (int r = 0; r < 8; ++r)
      t[(size_t)(mrow + r) * kH + ncol] = c[i][r];
  }
}

// ------ fused: agg = selfn*t + bias + sum_in enorm*t[src]; LN; ReLU; bf16 --
// One block (256 threads) per node; no atomics; LDS cross-wave LN reduction.
__global__ void k_fused(const float* __restrict__ t, const float* __restrict__ selfn,
                        const int* __restrict__ cnt, const int* __restrict__ adj,
                        const float* __restrict__ adjw, const float* __restrict__ cb,
                        const float* __restrict__ g, const float* __restrict__ bta,
                        unsigned short* __restrict__ hb) {
  __shared__ float red[16];
  int node = blockIdx.x;              // 0 .. kTotal-1
  int b = node >> 12;                 // kN = 4096
  int n = node & (kN - 1);
  int f = threadIdx.x;

  float acc = fmaf(selfn[n], t[((size_t)node << 8) + f], cb[f]);
  int c = min(cnt[n], kCap);
  for (int j = 0; j < c; ++j) {
    int s  = adj[n * kCap + j];
    float w = adjw[n * kCap + j];
    acc = fmaf(w, t[(((size_t)(b << 12) + (size_t)s) << 8) + f], acc);
  }

  // block-wide mean/var over 256 features
  float s1 = acc, s2 = acc * acc;
#pragma unroll
  for (int off = 16; off >= 1; off >>= 1) {
    s1 += __shfl_xor(s1, off, 32);
    s2 += __shfl_xor(s2, off, 32);
  }
  int wid = threadIdx.x >> 5, lane = threadIdx.x & 31;
  if (lane == 0) { red[wid] = s1; red[8 + wid] = s2; }
  __syncthreads();
  float sum = 0.f, sq = 0.f;
#pragma unroll
  for (int i = 0; i < 8; ++i) { sum += red[i]; sq += red[8 + i]; }

  float mu   = sum * (1.f / kH);
  float var  = sq * (1.f / kH) - mu * mu;
  float rinv = rsqrtf(var + kEps);
  float y = (acc - mu) * rinv * g[f] + bta[f];
  hb[((size_t)node << 8) + f] = f2bf(fmaxf(y, 0.f));
}

// ---------------- mean pool + MLP head -------------------------------------
__global__ void k_pool_zero(float* __restrict__ p) {
  p[blockIdx.x * blockDim.x + threadIdx.x] = 0.f;
}
__global__ void k_pool(const unsigned short* __restrict__ hb, float* __restrict__ pooled) {
  int b = blockIdx.x >> 4;
  int c = blockIdx.x & 15;           // 16 chunks of 256 nodes
  int f = threadIdx.x;
  const unsigned short* base = hb + ((((size_t)b * kN) + (size_t)c * 256) << 8) + f;
  float s = 0.f;
  for (int n = 0; n < 256; ++n) s += bf2f(base[(size_t)n << 8]);
  atomicAdd(&pooled[b * kH + f], s * (1.f / kN));
}
__global__ void k_mlp1(const float* __restrict__ pooled, const float* __restrict__ w,
                       const float* __restrict__ bia, float* __restrict__ mid) {
  int b = blockIdx.x, f = threadIdx.x;
  float acc = bia[f];
  for (int k = 0; k < kH; ++k) acc = fmaf(pooled[b * kH + k], w[k * kH + f], acc);
  mid[b * kH + f] = fmaxf(acc, 0.f);
}
__global__ void k_mlp2(const float* __restrict__ mid, const float* __restrict__ w,
                       const float* __restrict__ bia, float* __restrict__ out) {
  int b = blockIdx.x, o = threadIdx.x;
  float acc = bia[o];
  for (int k = 0; k < kH; ++k) acc = fmaf(mid[b * kH + k], w[k * kOut + o], acc);
  out[b * kOut + o] = acc;
}

}  // namespace

extern "C" void kernel_launch(void* const* d_in, const int* in_sizes, int n_in,
                              void* d_out, int out_size, void* d_ws, size_t ws_size,
                              hipStream_t stream) {
  const float* x      = (const float*)d_in[0];
  const int*   ei     = (const int*)d_in[1];
  const float* proj_w = (const float*)d_in[2];
  const float* proj_b = (const float*)d_in[3];
  const float* conv_w = (const float*)d_in[4];
  const float* conv_b = (const float*)d_in[5];
  const float* ln_g   = (const float*)d_in[6];
  const float* ln_b   = (const float*)d_in[7];
  const float* h1_w   = (const float*)d_in[8];
  const float* h1_b   = (const float*)d_in[9];
  const float* h2_w   = (const float*)d_in[10];
  const float* h2_b   = (const float*)d_in[11];
  float* out = (float*)d_out;

  const int E0 = in_sizes[1] / 2;      // edges per graph
  const int* src = ei;
  const int* dst = ei + E0;

  char* p = (char*)d_ws;
  auto alloc = [&](size_t bytes) {
    char* r = p;
    p += (bytes + 255) & ~(size_t)255;
    return r;
  };
  unsigned short* hb    = (unsigned short*)alloc((size_t)kTotal * kH * 2);  // 67 MB
  float*          t     = (float*)alloc((size_t)kTotal * kH * 4);           // 134 MB
  unsigned short* pw    = (unsigned short*)alloc((size_t)16 * 8 * 32 * 16 * 2);
  float*          deg   = (float*)alloc((size_t)kN * 4);
  float*          dinv  = (float*)alloc((size_t)kN * 4);
  float*          selfn = (float*)alloc((size_t)kN * 4);
  int*            cnt   = (int*)alloc((size_t)kN * 4);
  int*            adj   = (int*)alloc((size_t)kN * kCap * 4);
  float*          adjw  = (float*)alloc((size_t)kN * kCap * 4);
  float*          pooled= (float*)alloc((size_t)kB * kH * 4);
  float*          mid   = (float*)alloc((size_t)kB * kH * 4);

  // h0 = x * proj_w + proj_b  (bf16)
  k_proj<<<kTotal * kH / 4 / 256, 256, 0, stream>>>(x, proj_w, proj_b, hb);

  // degree / norms / adjacency (per-graph structure, shared across batch)
  k_deg_init<<<kN / 256, 256, 0, stream>>>(deg, cnt);
  k_deg_acc<<<(E0 + 255) / 256, 256, 0, stream>>>(dst, E0, deg);
  k_deg_fin<<<kN / 256, 256, 0, stream>>>(deg, dinv, selfn);
  k_adj_build<<<(E0 + 255) / 256, 256, 0, stream>>>(src, dst, dinv, E0, cnt, adj, adjw);

  for (int l = 0; l < kL; ++l) {
    k_pack_w<<<256, 256, 0, stream>>>(conv_w + (size_t)l * kH * kH, pw);
    k_gemm<<<(kTotal / 16) * 4 / 8, 256, 0, stream>>>(hb, pw, t);   // 4096 blocks
    k_fused<<<kTotal, 256, 0, stream>>>(t, selfn, cnt, adj, adjw,
                                        conv_b + l * kH, ln_g + l * kH,
                                        ln_b + l * kH, hb);
  }

  k_pool_zero<<<kB, 256, 0, stream>>>(pooled);
  k_pool<<<kB * 16, 256, 0, stream>>>(hb, pooled);
  k_mlp1<<<kB, 256, 0, stream>>>(pooled, h1_w, h1_b, mid);
  k_mlp2<<<kB, 256, 0, stream>>>(mid, h2_w, h2_b, out);
}
